// ExitBlock_22737556865408
// MI455X (gfx1250) — matrix operand order repaired and checked by reading, not compiled, not run on hardware
//
#include <hip/hip_runtime.h>
#include <hip/hip_bf16.h>

typedef __bf16 bf16_t;
typedef bf16_t bf16x4  __attribute__((ext_vector_type(4)));
typedef bf16_t bf16x8  __attribute__((ext_vector_type(8)));
typedef bf16_t bf16x16 __attribute__((ext_vector_type(16)));
typedef float  f32x4   __attribute__((ext_vector_type(4)));
typedef float  v8f     __attribute__((ext_vector_type(8)));

constexpr int Nn     = 12000;  // nodes
constexpr int NHID   = 32;
constexpr int NCLASS = 16;

constexpr int BM   = 64;        // rows per block (4 M-tiles x 16)
constexpr int BK   = 96;        // K slab per iteration (3 wmma K-steps); 12000 = 96*125
constexpr int APAD = 104;       // LDS row pitch (bf16): 208B, keeps b128 reads 16B-aligned
constexpr int BPAD = 104;
constexpr int KITERS = Nn / BK; // 125, exact

// ---------------------------------------------------------------------------
// async global->LDS copy of 8 bytes (gfx1250, ASYNCcnt-tracked)
// ---------------------------------------------------------------------------
__device__ __forceinline__ void async_copy_b64(unsigned lds_addr, const void* gptr) {
    asm volatile("global_load_async_to_lds_b64 %0, %1, off"
                 :
                 : "v"(lds_addr), "v"((unsigned long long)gptr)
                 : "memory");
}
__device__ __forceinline__ void wait_asynccnt0() {
    asm volatile("s_wait_asynccnt 0x0" ::: "memory");
}

// ---------------------------------------------------------------------------
// Kernel 1: support_T[j][n] = sum_k x[n][k] * gc_W[k][j]   (bf16, transposed)
// ---------------------------------------------------------------------------
__global__ __launch_bounds__(256)
void gcn_support_kernel(const float* __restrict__ x,
                        const float* __restrict__ gcW,
                        bf16_t* __restrict__ supT) {
    int id = blockIdx.x * blockDim.x + threadIdx.x;
    if (id >= Nn * NHID) return;
    int j = id / Nn;          // hidden index 0..31
    int n = id % Nn;          // node index (consecutive threads -> coalesced store)
    const float* xr = x + (size_t)n * NHID;
    float acc = 0.0f;
#pragma unroll
    for (int k = 0; k < NHID; ++k)
        acc += xr[k] * gcW[k * NHID + j];
    supT[(size_t)j * Nn + n] = (bf16_t)acc;
}

// ---------------------------------------------------------------------------
// Kernel 2: H = adj @ support   (M=12000, K=12000, N=32), bf16 WMMA, f32 acc
//   - 8 waves: wave = (Mtile 0..3) x (Ntile 0..1), one 16x16 tile per wave
//   - A slab (adj, f32) register-staged, v_cvt_pk_bf16_f32 -> LDS bf16
//   - B slab (supT, bf16) copied with global_load_async_to_lds_b64
// ---------------------------------------------------------------------------
__global__ __launch_bounds__(256)
void gcn_adj_matmul_kernel(const float* __restrict__ adj,
                           const bf16_t* __restrict__ supT,
                           float* __restrict__ H) {
    __shared__ bf16_t Atile[BM * APAD];     // 64 x 96 (pitch 104) bf16 = 13312 B
    __shared__ bf16_t Btile[NHID * BPAD];   // 32 x 96 (pitch 104) bf16 =  6656 B

    const int tid   = threadIdx.x;
    const int lane  = tid & 31;
    const int wave  = tid >> 5;             // 0..7
    const int mtile = wave >> 1;            // 0..3
    const int ntile = wave & 1;             // 0..1
    const int lhalf = lane >> 4;            // 0/1
    const int lmod  = lane & 15;
    const int r0    = blockIdx.x * BM;

    // ---- A loader mapping: one row / thread, 6 f32x4 chunks across K ----
    const int arow    = tid >> 2;           // 0..63
    const int acolb   = (tid & 3) * 4;      // 0,4,8,12 ; chunk i at acolb + 16*i
    const int garow   = r0 + arow;
    const size_t growA = (size_t)(garow < Nn ? garow : (Nn - 1));

    // ---- B loader mapping: 3 b64 async copies / thread ----
    const int bn    = tid >> 3;             // 0..31  (hidden index)
    const int bcolb = (tid & 7) * 4;        // 0..28 ; chunk i at bcolb + 32*i

    // ---- stage slab 0 ----
    f32x4 areg[6];
#pragma unroll
    for (int i = 0; i < 6; ++i)
        areg[i] = *(const f32x4*)(adj + growA * (size_t)Nn + acolb + 16 * i);

    // Fragment LDS offsets (bf16 elems), 16-bit WMMA striping:
    //  A (16x32): lanes0-15 K in {0..7,16..23}, lanes16-31 K in {8..15,24..31}
    //  B (32x16): lanes0-15 K in {0..7,16..23}, lanes16-31 K in {8..15,24..31}
    const int aoff = (mtile * 16 + lmod) * APAD + 8 * lhalf;
    const int boff = (ntile * 16 + lmod) * BPAD + 8 * lhalf;

    v8f acc = {};

    for (int it = 0; it < KITERS; ++it) {
        const int k0 = it * BK;
        __syncthreads();                    // previous slab fully consumed

        // B slab: async copy global bf16 -> LDS (no conversion needed)
#pragma unroll
        for (int i = 0; i < 3; ++i) {
            const int c = bcolb + 32 * i;
            async_copy_b64((unsigned)(unsigned long long)&Btile[bn * BPAD + c],
                           supT + (size_t)bn * Nn + k0 + c);
        }

        // A slab: registers -> bf16 -> LDS
#pragma unroll
        for (int i = 0; i < 6; ++i) {
            bf16x4 v;
#pragma unroll
            for (int j = 0; j < 4; ++j) v[j] = (bf16_t)areg[i][j];
            *(bf16x4*)&Atile[arow * APAD + acolb + 16 * i] = v;
        }

        // stage next A slab while matrix pipes chew on this one
        if (it + 1 < KITERS) {
            const int k0n = k0 + BK;
#pragma unroll
            for (int i = 0; i < 6; ++i)
                areg[i] = *(const f32x4*)(adj + growA * (size_t)Nn + k0n + acolb + 16 * i);
            if (it + 2 < KITERS)
                __builtin_prefetch(adj + growA * (size_t)Nn + k0 + 2 * BK + acolb, 0, 1);
        }

        wait_asynccnt0();                   // our async B writes landed in LDS
        __syncthreads();                    // everyone's A+B slab visible

        // 3 WMMA K-steps over the 96-deep slab
#pragma unroll
        for (int s = 0; s < 3; ++s) {
            bf16x8 a_lo = *(const bf16x8*)&Atile[aoff + 32 * s];
            bf16x8 a_hi = *(const bf16x8*)&Atile[aoff + 32 * s + 16];
            bf16x8 b_lo = *(const bf16x8*)&Btile[boff + 32 * s];
            bf16x8 b_hi = *(const bf16x8*)&Btile[boff + 32 * s + 16];

            bf16x16 afrag, bfrag;
#pragma unroll
            for (int j = 0; j < 8; ++j) {
                afrag[j] = a_lo[j]; afrag[8 + j] = a_hi[j];
                bfrag[j] = b_lo[j]; bfrag[8 + j] = b_hi[j];
            }
            acc = __builtin_amdgcn_wmma_f32_16x16x32_bf16(
                      false, afrag, false, bfrag, (short)0, acc, false, false);
        }
    }

    // C/D layout: VGPR r -> M = r + 8*(lane>=16), N = lane%16
#pragma unroll
    for (int r = 0; r < 8; ++r) {
        int m = mtile * 16 + r + 8 * lhalf;
        int g = r0 + m;
        if (g < Nn)
            H[(size_t)g * NHID + ntile * 16 + lmod] = acc[r];
    }
}

// ---------------------------------------------------------------------------
// Kernel 3: out = (H + gc_b) @ fc_W + fc_b
// ---------------------------------------------------------------------------
__global__ __launch_bounds__(256)
void gcn_head_kernel(const float* __restrict__ H,
                     const float* __restrict__ gcb,
                     const float* __restrict__ fcW,
                     const float* __restrict__ fcb,
                     float* __restrict__ out) {
    int id = blockIdx.x * blockDim.x + threadIdx.x;
    if (id >= Nn * NCLASS) return;
    int n = id / NCLASS;
    int c = id % NCLASS;
    const float* hr = H + (size_t)n * NHID;
    float acc = fcb[c];
#pragma unroll
    for (int k = 0; k < NHID; ++k)
        acc += (hr[k] + gcb[k]) * fcW[k * NCLASS + c];
    out[id] = acc;
}

// ---------------------------------------------------------------------------
extern "C" void kernel_launch(void* const* d_in, const int* in_sizes, int n_in,
                              void* d_out, int out_size, void* d_ws, size_t ws_size,
                              hipStream_t stream) {
    const float* x   = (const float*)d_in[0];
    const float* adj = (const float*)d_in[1];
    const float* gcW = (const float*)d_in[2];
    const float* gcb = (const float*)d_in[3];
    const float* fcW = (const float*)d_in[4];
    const float* fcb = (const float*)d_in[5];
    float* out = (float*)d_out;

    // workspace layout
    bf16_t* supT = (bf16_t*)d_ws;                              // 32*12000*2 = 768000 B
    float*  H    = (float*)((char*)d_ws + (size_t)768000);     // 12000*32*4 = 1536000 B

    gcn_support_kernel<<<(Nn * NHID + 255) / 256, 256, 0, stream>>>(x, gcW, supT);
    gcn_adj_matmul_kernel<<<(Nn + BM - 1) / BM, 256, 0, stream>>>(adj, supT, H);
    gcn_head_kernel<<<(Nn * NCLASS + 255) / 256, 256, 0, stream>>>(H, gcb, fcW, fcb, out);
}
